// GPT2Attention_13151189860910
// MI455X (gfx1250) — matrix operand order repaired and checked by reading, not compiled, not run on hardware
//
#include <hip/hip_runtime.h>

#define BQ     8
#define LSEQ   1024
#define DEMB   768
#define NHEAD  12
#define HDIM   64
#define MTOT   (BQ * LSEQ)     // 8192
#define NQKV   (3 * DEMB)      // 2304

typedef __attribute__((ext_vector_type(16))) _Float16 v16h;
typedef __attribute__((ext_vector_type(8)))  _Float16 v8h;
typedef __attribute__((ext_vector_type(8)))  float    v8f;

// ---------------------------------------------------------------------------
// WMMA fragment loaders for V_WMMA_F32_16X16X32_F16 (wave32), per ISA 7.12.2.
// Layouts chosen so each lane's fragment is two contiguous 16B runs
// (pure ds_load_b128, no half-register repacking).
//
//  A (16x32): lane half g=lane>=16. a[0..7]=K[g*8..g*8+7], a[8..15]=K[16+g*8..23+g*8]
//             -> row-major tile: two 16B loads at +g*8 and +16+g*8.
//  B (32x16): b[0..7]=K[g*8..g*8+7], b[8..15]=K[16+g*8..23+g*8], N=lane&15 -> K-innermost (transposed)
//             tile: two 16B loads at (n)*ld + koff + g*8 and +16.
// ---------------------------------------------------------------------------
__device__ __forceinline__ v16h frag_a(const _Float16* __restrict__ s,
                                       int row0, int ld, int koff) {
  const int lane = threadIdx.x & 31;
  const int g = (lane >> 4) & 1;
  const int m = lane & 15;
  const _Float16* p = s + (row0 + m) * ld + koff + g * 8;
  const v8h lo = *(const v8h*)(p);
  const v8h hi = *(const v8h*)(p + 16);
  return __builtin_shufflevector(lo, hi, 0, 1, 2, 3, 4, 5, 6, 7,
                                 8, 9, 10, 11, 12, 13, 14, 15);
}

// Logical B[k][n] stored K-innermost: element (k,n) at s[(n0+n)*ld + koff + k].
__device__ __forceinline__ v16h frag_bk(const _Float16* __restrict__ s,
                                        int n0, int ld, int koff) {
  const int lane = threadIdx.x & 31;
  const int g = (lane >> 4) & 1;
  const int n = lane & 15;
  const _Float16* p = s + (n0 + n) * ld + koff + g * 8;
  const v8h lo = *(const v8h*)(p);
  const v8h hi = *(const v8h*)(p + 16);
  return __builtin_shufflevector(lo, hi, 0, 1, 2, 3, 4, 5, 6, 7,
                                 8, 9, 10, 11, 12, 13, 14, 15);
}

#define WMMA_F32F16(acc, a, b)                                              \
  (acc) = __builtin_amdgcn_wmma_f32_16x16x32_f16(false, (a), false, (b),    \
                                                 (short)0, (acc), false, false)

// ---------------------------------------------------------------------------
// Kernel 1: qkv = x @ w_attn + b_attn ; scatter q/k/v as f16 [b,h,l,hd]
// block tile 128(M) x 64(N); 8 waves -> 32x32 quadrants; K step 32
// ---------------------------------------------------------------------------
__global__ __launch_bounds__(256, 2) void qkv_gemm_kernel(
    const float* __restrict__ x, const float* __restrict__ w,
    const float* __restrict__ bias, _Float16* __restrict__ qo,
    _Float16* __restrict__ ko, _Float16* __restrict__ vo) {
  __shared__ _Float16 As[128][40];   // row-major M x K
  __shared__ _Float16 Bt[64][40];    // K-innermost: [n][k]
  const int tid  = threadIdx.x;
  const int lane = tid & 31;
  const int wv   = tid >> 5;
  const int wm   = wv & 3;
  const int wn   = wv >> 2;
  const int row0 = blockIdx.y * 128;
  const int col0 = blockIdx.x * 64;

  v8f acc[2][2] = {};
  for (int kt = 0; kt < DEMB / 32; ++kt) {
    const int k0 = kt * 32;
    // stage A: 128x32 fp32 -> f16 (row-major)
#pragma unroll
    for (int i = 0; i < 4; ++i) {
      const int idx = tid + i * 256;
      const int r = idx >> 3, c = (idx & 7) * 4;
      const float* src = x + (size_t)(row0 + r) * DEMB + k0 + c;
      if (kt + 1 < DEMB / 32) __builtin_prefetch(src + 32, 0, 1);
      const float4 f = *(const float4*)src;
      As[r][c + 0] = (_Float16)f.x; As[r][c + 1] = (_Float16)f.y;
      As[r][c + 2] = (_Float16)f.z; As[r][c + 3] = (_Float16)f.w;
    }
    // stage B: 32x64 fp32 -> f16, transposed into [n][k]
#pragma unroll
    for (int i = 0; i < 2; ++i) {
      const int idx = tid + i * 256;
      const int r = idx >> 4, c = (idx & 15) * 4;   // r = k row, c = n base
      const float4 f = *(const float4*)(w + (size_t)(k0 + r) * NQKV + col0 + c);
      Bt[c + 0][r] = (_Float16)f.x; Bt[c + 1][r] = (_Float16)f.y;
      Bt[c + 2][r] = (_Float16)f.z; Bt[c + 3][r] = (_Float16)f.w;
    }
    __syncthreads();
    v16h a0 = frag_a(&As[0][0], wm * 32,      40, 0);
    v16h a1 = frag_a(&As[0][0], wm * 32 + 16, 40, 0);
    v16h b0 = frag_bk(&Bt[0][0], wn * 32,      40, 0);
    v16h b1 = frag_bk(&Bt[0][0], wn * 32 + 16, 40, 0);
    WMMA_F32F16(acc[0][0], a0, b0);
    WMMA_F32F16(acc[0][1], a0, b1);
    WMMA_F32F16(acc[1][0], a1, b0);
    WMMA_F32F16(acc[1][1], a1, b1);
    __syncthreads();
  }

  const int g = (lane >> 4) & 1;
  const int n15 = lane & 15;
#pragma unroll
  for (int i = 0; i < 2; ++i)
#pragma unroll
    for (int j = 0; j < 2; ++j)
#pragma unroll
      for (int e = 0; e < 8; ++e) {
        const int gr = row0 + wm * 32 + i * 16 + g * 8 + e;
        const int gc = col0 + wn * 32 + j * 16 + n15;
        const float val = acc[i][j][e] + bias[gc];
        const int bb = gr >> 10, ll = gr & 1023;
        const int t  = gc / DEMB;
        const int dd = gc - t * DEMB;
        const int hh = dd >> 6, hp = dd & 63;
        _Float16* dst = (t == 0) ? qo : (t == 1) ? ko : vo;
        dst[(((size_t)(bb * NHEAD + hh)) * LSEQ + ll) * HDIM + hp] = (_Float16)val;
      }
}

// ---------------------------------------------------------------------------
// Kernel 2: flash attention, one block per (b, h, 64-query tile).
// Stores O with the reference's faithful swapaxes(1,3) scramble as f16.
// ---------------------------------------------------------------------------
__global__ __launch_bounds__(256, 2) void attn_kernel(
    const _Float16* __restrict__ q, const _Float16* __restrict__ k,
    const _Float16* __restrict__ v, _Float16* __restrict__ oscr) {
  __shared__ _Float16 Qs[64][72];    // [query][d]      (A side, row-major)
  __shared__ _Float16 Ks[64][72];    // [key][d]        (B side of QK^T: K-innermost)
  __shared__ _Float16 Vt[64][72];    // [hd][key]       (B side of P@V: K-innermost)
  __shared__ float    Sb[64][66];
  __shared__ _Float16 Pb[64][72];    // [query][key]    (A side of P@V)
  __shared__ float    s_alpha[64];
  __shared__ float    s_m[64];
  __shared__ float    s_l[64];

  const int tid  = threadIdx.x;
  const int lane = tid & 31;
  const int wv   = tid >> 5;
  const int qt   = blockIdx.x & 15;   // query tile (of 16)
  const int bh   = blockIdx.x >> 4;   // b*12 + h
  const int hh   = bh % NHEAD;
  const int bb   = bh / NHEAD;

  const size_t base = (size_t)bh * LSEQ * HDIM;
  const _Float16* qp = q + base + (size_t)qt * 64 * HDIM;
  const _Float16* kp = k + base;
  const _Float16* vp = v + base;

  // load 64x64 Q tile (contiguous)
#pragma unroll
  for (int i = 0; i < 2; ++i) {
    const int idx = tid + i * 256;
    const int r = idx >> 3, c = (idx & 7) * 8;
    *(uint4*)&Qs[r][c] = *(const uint4*)(qp + (size_t)idx * 8);
  }
  if (tid < 64) { s_m[tid] = -3.0e38f; s_l[tid] = 0.0f; }
  __syncthreads();

  const int g   = (lane >> 4) & 1;
  const int n15 = lane & 15;
  const int qb  = (wv >> 1) * 16;        // wave's 16 query rows
  const int nb0 = ((wv & 1) * 2) * 16;   // wave's 2 column tiles: nb0, nb0+16
  v8f oacc[2] = {};

  for (int kb = 0; kb < LSEQ; kb += 64) {
    // load K tile row-major; V tile transposed into [hd][key]
#pragma unroll
    for (int i = 0; i < 2; ++i) {
      const int idx = tid + i * 256;
      const int r = idx >> 3, c = (idx & 7) * 8;   // r = key row, c = hd base
      *(uint4*)&Ks[r][c] = *(const uint4*)(kp + (size_t)kb * HDIM + (size_t)idx * 8);
      union { uint4 u; _Float16 h[8]; } tv;
      tv.u = *(const uint4*)(vp + (size_t)kb * HDIM + (size_t)idx * 8);
#pragma unroll
      for (int t = 0; t < 8; ++t) Vt[c + t][r] = tv.h[t];
    }
    __syncthreads();

    // S = Q K^T over hd=64 (2 WMMA k-steps), wave covers 16x32 of 64x64 S
    v8f sacc[2] = {};
#pragma unroll
    for (int ks = 0; ks < HDIM; ks += 32) {
      v16h aq  = frag_a(&Qs[0][0], qb, 72, ks);
      v16h bk0 = frag_bk(&Ks[0][0], nb0,      72, ks);
      v16h bk1 = frag_bk(&Ks[0][0], nb0 + 16, 72, ks);
      WMMA_F32F16(sacc[0], aq, bk0);
      WMMA_F32F16(sacc[1], aq, bk1);
    }
#pragma unroll
    for (int j = 0; j < 2; ++j)
#pragma unroll
      for (int e = 0; e < 8; ++e)
        Sb[qb + g * 8 + e][nb0 + j * 16 + n15] = sacc[j][e] * 0.125f;
    __syncthreads();

    // online softmax: one thread per query row
    if (tid < 64) {
      const int r = tid;
      const float mold = s_m[r];
      float mx = mold;
#pragma unroll 8
      for (int c = 0; c < 64; ++c) mx = fmaxf(mx, Sb[r][c]);
      const float al = __expf(mold - mx);
      float sum = 0.0f;
#pragma unroll 8
      for (int c = 0; c < 64; ++c) {
        const float p = __expf(Sb[r][c] - mx);
        Pb[r][c] = (_Float16)p;
        sum += p;
      }
      s_m[r] = mx;
      s_l[r] = s_l[r] * al + sum;
      s_alpha[r] = al;
    }
    __syncthreads();

    // rescale O, then O += P @ V (2 WMMA k-steps over the 64 keys)
#pragma unroll
    for (int j = 0; j < 2; ++j)
#pragma unroll
      for (int e = 0; e < 8; ++e)
        oacc[j][e] *= s_alpha[qb + g * 8 + e];
#pragma unroll
    for (int ks = 0; ks < 64; ks += 32) {
      v16h ap  = frag_a(&Pb[0][0], qb, 72, ks);
      v16h bv0 = frag_bk(&Vt[0][0], nb0,      72, ks);
      v16h bv1 = frag_bk(&Vt[0][0], nb0 + 16, 72, ks);
      WMMA_F32F16(oacc[0], ap, bv0);
      WMMA_F32F16(oacc[1], ap, bv1);
    }
    __syncthreads();
  }

  // normalize and store with faithful swapaxes(1,3) scramble:
  // within batch: flat index f = hd*(L*H) + l*H + h
#pragma unroll
  for (int j = 0; j < 2; ++j)
#pragma unroll
    for (int e = 0; e < 8; ++e) {
      const int qr = qb + g * 8 + e;
      const int lq = qt * 64 + qr;
      const int hd = nb0 + j * 16 + n15;
      const float val = oacc[j][e] / s_l[qr];
      const size_t f = (size_t)hd * (LSEQ * NHEAD) + (size_t)lq * NHEAD + hh;
      oscr[(size_t)bb * (LSEQ * DEMB) + f] = (_Float16)val;
    }
}

// ---------------------------------------------------------------------------
// Kernel 3: out = oscr @ w_proj + b_proj  (fp32 output)
// ---------------------------------------------------------------------------
__global__ __launch_bounds__(256, 2) void proj_gemm_kernel(
    const _Float16* __restrict__ a, const float* __restrict__ w,
    const float* __restrict__ bias, float* __restrict__ out) {
  __shared__ _Float16 As[128][40];   // row-major M x K
  __shared__ _Float16 Bt[64][40];    // K-innermost: [n][k]
  const int tid  = threadIdx.x;
  const int lane = tid & 31;
  const int wv   = tid >> 5;
  const int wm   = wv & 3;
  const int wn   = wv >> 2;
  const int row0 = blockIdx.y * 128;
  const int col0 = blockIdx.x * 64;

  v8f acc[2][2] = {};
  for (int kt = 0; kt < DEMB / 32; ++kt) {
    const int k0 = kt * 32;
#pragma unroll
    for (int i = 0; i < 2; ++i) {   // A already f16: 512 x 16B chunks
      const int idx = tid + i * 256;
      const int r = idx >> 2, c = (idx & 3) * 8;
      *(uint4*)&As[r][c] = *(const uint4*)(a + (size_t)(row0 + r) * DEMB + k0 + c);
    }
#pragma unroll
    for (int i = 0; i < 2; ++i) {   // B fp32 -> f16, transposed into [n][k]
      const int idx = tid + i * 256;
      const int r = idx >> 4, c = (idx & 15) * 4;   // r = k row, c = n base
      const float4 f = *(const float4*)(w + (size_t)(k0 + r) * DEMB + col0 + c);
      Bt[c + 0][r] = (_Float16)f.x; Bt[c + 1][r] = (_Float16)f.y;
      Bt[c + 2][r] = (_Float16)f.z; Bt[c + 3][r] = (_Float16)f.w;
    }
    __syncthreads();
    v16h a0 = frag_a(&As[0][0], wm * 32,      40, 0);
    v16h a1 = frag_a(&As[0][0], wm * 32 + 16, 40, 0);
    v16h b0 = frag_bk(&Bt[0][0], wn * 32,      40, 0);
    v16h b1 = frag_bk(&Bt[0][0], wn * 32 + 16, 40, 0);
    WMMA_F32F16(acc[0][0], a0, b0);
    WMMA_F32F16(acc[0][1], a0, b1);
    WMMA_F32F16(acc[1][0], a1, b0);
    WMMA_F32F16(acc[1][1], a1, b1);
    __syncthreads();
  }

  const int g = (lane >> 4) & 1;
  const int n15 = lane & 15;
#pragma unroll
  for (int i = 0; i < 2; ++i)
#pragma unroll
    for (int j = 0; j < 2; ++j)
#pragma unroll
      for (int e = 0; e < 8; ++e) {
        const int gr = row0 + wm * 32 + i * 16 + g * 8 + e;
        const int gc = col0 + wn * 32 + j * 16 + n15;
        out[(size_t)gr * DEMB + gc] = acc[i][j][e] + bias[gc];
      }
}

// ---------------------------------------------------------------------------
extern "C" void kernel_launch(void* const* d_in, const int* in_sizes, int n_in,
                              void* d_out, int out_size, void* d_ws, size_t ws_size,
                              hipStream_t stream) {
  (void)in_sizes; (void)n_in; (void)out_size; (void)ws_size;
  const float* x      = (const float*)d_in[0];
  const float* w_attn = (const float*)d_in[1];
  const float* b_attn = (const float*)d_in[2];
  const float* w_proj = (const float*)d_in[3];
  const float* b_proj = (const float*)d_in[4];
  float* out = (float*)d_out;

  const size_t elems = (size_t)BQ * NHEAD * LSEQ * HDIM;  // 6,291,456 per tensor
  _Float16* qws  = (_Float16*)d_ws;
  _Float16* kws  = qws + elems;
  _Float16* vws  = kws + elems;
  _Float16* oscr = vws + elems;  // [b][1024][768] scrambled f16

  qkv_gemm_kernel<<<dim3(NQKV / 64, MTOT / 128), 256, 0, stream>>>(
      x, w_attn, b_attn, qws, kws, vws);
  attn_kernel<<<dim3(BQ * NHEAD * (LSEQ / 64)), 256, 0, stream>>>(
      qws, kws, vws, oscr);
  proj_gemm_kernel<<<dim3(DEMB / 64, MTOT / 128), 256, 0, stream>>>(
      oscr, w_proj, b_proj, out);
}
